// GraphAutoEncoder_7112465842784
// MI455X (gfx1250) — matrix operand order repaired and checked by reading, not compiled, not run on hardware
//
#include <hip/hip_runtime.h>
#include <hip/hip_bf16.h>

// ---------------- problem constants ----------------
#define B_  32
#define N_  2048
#define E_  32768
#define ET_ (E_ + N_)        // edges + self loops = 34816
#define H_  64
#define ALPHA_ 0.1f

#define NTH_   (B_ * N_ * H_)     // 4,194,304
#define NN_    (B_ * N_)          // 65,536
#define EDG_   (B_ * ET_)         // 1,114,112
#define TILES_ (B_ * (N_/16) * (H_/16))   // 16384 wmma tiles per 64x64 GEMM

typedef __attribute__((ext_vector_type(16))) _Float16 v16h;
typedef __attribute__((ext_vector_type(8)))  float    v8f;

// ---------------- ordered-uint encoding for float atomicMax ----------------
__device__ __forceinline__ unsigned int enc_f(float f) {
  unsigned int u = __float_as_uint(f);
  return (u & 0x80000000u) ? ~u : (u | 0x80000000u);
}
__device__ __forceinline__ float dec_f(unsigned int u) {
  unsigned int v = (u & 0x80000000u) ? (u & 0x7FFFFFFFu) : ~u;
  return __uint_as_float(v);
}

// ---------------- WMMA fragment loaders (ISA 7.12.2 wave32 layouts) ----------------
// A: 16x32 f16 tile. lane L: M = L%16. For lanes 0-15: VGPR0-3 hold K=0..7,
// VGPR4-7 hold K=16..23; lanes 16-31 get +8 on those K bases.
__device__ __forceinline__ v16h load_a64(const float* xrow, int lane, int kk) {
  v16h a;
  int khi = (lane >> 4) << 3;   // 0 or 8
#pragma unroll
  for (int v = 0; v < 8; ++v) {
    int kb = kk + ((v < 4) ? (2 * v + khi) : (16 + 2 * (v - 4) + khi));
    a[2 * v]     = (_Float16)xrow[kb];
    a[2 * v + 1] = (_Float16)xrow[kb + 1];
  }
  return a;
}
// B: 32x16 f16 tile of row-major W (64x64). lane L: N = n (passed absolute),
// K = same order as A: v<4 -> kk + 2v + 8*(L/16), v>=4 -> kk + 16 + 2(v-4) + 8*(L/16).
__device__ __forceinline__ v16h load_b64(const float* W, int n, int lane, int kk) {
  v16h b;
  int khi = (lane >> 4) << 3;   // 0 or 8
#pragma unroll
  for (int v = 0; v < 8; ++v) {
    int k = kk + ((v < 4) ? (2 * v + khi) : (16 + 2 * (v - 4) + khi));
    b[2 * v]     = (_Float16)W[k * H_ + n];
    b[2 * v + 1] = (_Float16)W[(k + 1) * H_ + n];
  }
  return b;
}

// ---------------- GEMM: Y(N x 64) = X(N x 64) @ W(64x64) + bias, optional relu ----------------
__global__ void __launch_bounds__(256) gemm64_k(const float* __restrict__ X,
                                                const float* __restrict__ W,
                                                const float* __restrict__ bias,
                                                float* __restrict__ Y, int relu) {
  int wave = threadIdx.x >> 5, lane = threadIdx.x & 31;
  int t = blockIdx.x * 8 + wave;          // exact grid: TILES_/8 blocks
  int nt = t & 3; t >>= 2;
  int mt = t & 127; int b = t >> 7;
  const float* xrow = X + ((b * N_ + mt * 16) + (lane & 15)) * H_;
  v16h a0 = load_a64(xrow, lane, 0);
  v16h a1 = load_a64(xrow, lane, 32);
  int n = nt * 16 + (lane & 15);
  v16h b0 = load_b64(W, n, lane, 0);
  v16h b1 = load_b64(W, n, lane, 32);
  v8f c = {};
  c = __builtin_amdgcn_wmma_f32_16x16x32_f16(false, a0, false, b0, (short)0, c, false, false);
  c = __builtin_amdgcn_wmma_f32_16x16x32_f16(false, a1, false, b1, (short)0, c, false, false);
  int rhi = (lane >> 4) << 3;
  float bv = bias[n];
#pragma unroll
  for (int v = 0; v < 8; ++v) {
    int row = mt * 16 + v + rhi;
    float val = c[v] + bv;
    if (relu) val = fmaxf(val, 0.f);
    Y[(b * N_ + row) * H_ + n] = val;
  }
}

// ---------------- dual GEMM for GATv2 projections: xl = X@Wl+bl, xr = X@Wr+br ----------------
__global__ void __launch_bounds__(256) xlxr64_k(const float* __restrict__ X,
                                                const float* __restrict__ Wl,
                                                const float* __restrict__ bl,
                                                const float* __restrict__ Wr,
                                                const float* __restrict__ br,
                                                float* __restrict__ XL,
                                                float* __restrict__ XR) {
  int wave = threadIdx.x >> 5, lane = threadIdx.x & 31;
  int t = blockIdx.x * 8 + wave;
  int nt = t & 3; t >>= 2;
  int mt = t & 127; int b = t >> 7;
  const float* xrow = X + ((b * N_ + mt * 16) + (lane & 15)) * H_;
  v16h a0 = load_a64(xrow, lane, 0);
  v16h a1 = load_a64(xrow, lane, 32);
  int n = nt * 16 + (lane & 15);
  int rhi = (lane >> 4) << 3;

  v16h b0 = load_b64(Wl, n, lane, 0);
  v16h b1 = load_b64(Wl, n, lane, 32);
  v8f c = {};
  c = __builtin_amdgcn_wmma_f32_16x16x32_f16(false, a0, false, b0, (short)0, c, false, false);
  c = __builtin_amdgcn_wmma_f32_16x16x32_f16(false, a1, false, b1, (short)0, c, false, false);
  float bv = bl[n];
#pragma unroll
  for (int v = 0; v < 8; ++v) {
    int row = mt * 16 + v + rhi;
    XL[(b * N_ + row) * H_ + n] = c[v] + bv;
  }

  v16h b2 = load_b64(Wr, n, lane, 0);
  v16h b3 = load_b64(Wr, n, lane, 32);
  v8f c2 = {};
  c2 = __builtin_amdgcn_wmma_f32_16x16x32_f16(false, a0, false, b2, (short)0, c2, false, false);
  c2 = __builtin_amdgcn_wmma_f32_16x16x32_f16(false, a1, false, b3, (short)0, c2, false, false);
  float bv2 = br[n];
#pragma unroll
  for (int v = 0; v < 8; ++v) {
    int row = mt * 16 + v + rhi;
    XR[(b * N_ + row) * H_ + n] = c2[v] + bv2;
  }
}

// ---------------- pass-through outputs: obs[:,:,:4], obs[:,:,4:5] ----------------
__global__ void __launch_bounds__(256) copy_obs_k(const float* __restrict__ obs,
                                                  float* __restrict__ o0,
                                                  float* __restrict__ o1) {
  int i = blockIdx.x * 256 + threadIdx.x;
  if (i >= NN_) return;
  const float* p = obs + i * 5;
#pragma unroll
  for (int j = 0; j < 4; ++j) o0[i * 4 + j] = p[j];
  o1[i] = p[4];
}

// ---------------- encoder layer 1: h1 = relu(obs @ w1 + b1), 5 -> 64 ----------------
__global__ void __launch_bounds__(256) enc1_k(const float* __restrict__ obs,
                                              const float* __restrict__ w1,
                                              const float* __restrict__ b1,
                                              float* __restrict__ h1) {
  int i = blockIdx.x * 256 + threadIdx.x;
  if (i >= NTH_) return;
  int h = i & 63, bn = i >> 6;
  const float* p = obs + bn * 5;
  float acc = b1[h];
#pragma unroll
  for (int k = 0; k < 5; ++k) acc += p[k] * w1[k * H_ + h];
  h1[i] = fmaxf(acc, 0.f);
}

// ---------------- encoder layer 3: latent = h2 @ w3 + b3, 64 -> 3 ----------------
__global__ void __launch_bounds__(256) enc3_k(const float* __restrict__ h2,
                                              const float* __restrict__ w3,
                                              const float* __restrict__ b3,
                                              float* __restrict__ lat,
                                              float* __restrict__ outLat,
                                              float* __restrict__ xg) {
  int i = blockIdx.x * 256 + threadIdx.x;
  if (i >= NN_) return;
  const float* hp = h2 + i * H_;
#pragma unroll
  for (int j = 0; j < 3; ++j) {
    float acc = b3[j];
    for (int k = 0; k < H_; ++k) acc += hp[k] * w3[k * 3 + j];
    lat[i * 3 + j] = acc;
    outLat[i * 3 + j] = acc;
    if (j == 2) xg[i] = acc;
  }
}

// ---------------- skip: sk = ALPHA*(latent @ wsk + bsk), 3 -> 64 ----------------
__global__ void __launch_bounds__(256) skip_k(const float* __restrict__ lat,
                                              const float* __restrict__ wsk,
                                              const float* __restrict__ bsk,
                                              float* __restrict__ sk) {
  int i = blockIdx.x * 256 + threadIdx.x;
  if (i >= NTH_) return;
  int h = i & 63, bn = i >> 6;
  const float* lp = lat + bn * 3;
  float acc = bsk[h];
#pragma unroll
  for (int k = 0; k < 3; ++k) acc += lp[k] * wsk[k * H_ + h];
  sk[i] = ALPHA_ * acc;
}

// ---------------- GAT1 projections (fin = 1, outer product) ----------------
__global__ void __launch_bounds__(256) lin1_k(const float* __restrict__ xg,
                                              const float* __restrict__ Wl,
                                              const float* __restrict__ bl,
                                              const float* __restrict__ Wr,
                                              const float* __restrict__ br,
                                              float* __restrict__ XL,
                                              float* __restrict__ XR) {
  int i = blockIdx.x * 256 + threadIdx.x;
  if (i >= NTH_) return;
  int h = i & 63, bn = i >> 6;
  float x = xg[bn];
  XL[i] = x * Wl[h] + bl[h];
  XR[i] = x * Wr[h] + br[h];
}

// ---------------- segment init / zero ----------------
__global__ void __launch_bounds__(256) initnd_k(unsigned int* __restrict__ mb,
                                                float* __restrict__ den) {
  int i = blockIdx.x * 256 + threadIdx.x;
  if (i >= NN_) return;
  mb[i] = 0u;           // identity for ordered-uint max
  den[i] = 0.f;
}
__global__ void __launch_bounds__(256) zero_k(float* __restrict__ p, int n) {
  int i = blockIdx.x * 256 + threadIdx.x;
  if (i < n) p[i] = 0.f;
}

// ---------------- edge pass 1: score + segment_max (one wave per edge) ----------------
__global__ void __launch_bounds__(256) edge_score_k(const float* __restrict__ XL,
                                                    const float* __restrict__ XR,
                                                    const int* __restrict__ ei,
                                                    const float* __restrict__ att,
                                                    float* __restrict__ score,
                                                    unsigned int* __restrict__ mb) {
  int wave = threadIdx.x >> 5, lane = threadIdx.x & 31;
  int w = blockIdx.x * 8 + wave;
  if (w >= EDG_) return;
  int b = w / ET_, e = w % ET_;
  int s, d;
  if (e < E_) { s = ei[b * 2 * E_ + e]; d = ei[b * 2 * E_ + E_ + e]; }
  else        { s = d = e - E_; }
  const float* pl = XL + (b * N_ + s) * H_;
  const float* pr = XR + (b * N_ + d) * H_;
  float acc = 0.f;
#pragma unroll
  for (int h = lane; h < H_; h += 32) {
    float v = pl[h] + pr[h];
    v = v > 0.f ? v : 0.2f * v;          // leaky_relu(., 0.2)
    acc += att[h] * v;
  }
#pragma unroll
  for (int o = 16; o; o >>= 1) acc += __shfl_xor(acc, o, 32);
  if (lane == 0) {
    score[w] = acc;
    atomicMax(&mb[b * N_ + d], enc_f(acc));
  }
}

// ---------------- edge pass 2: ez = exp(score - m[dst]); denom[dst] += ez ----------------
__global__ void __launch_bounds__(256) edge_soft_k(float* __restrict__ score,
                                                   const unsigned int* __restrict__ mb,
                                                   float* __restrict__ den,
                                                   const int* __restrict__ ei) {
  int i = blockIdx.x * 256 + threadIdx.x;
  if (i >= EDG_) return;
  int b = i / ET_, e = i % ET_;
  int d = (e < E_) ? ei[b * 2 * E_ + E_ + e] : (e - E_);
  float m = dec_f(mb[b * N_ + d]);
  float ez = __expf(score[i] - m);
  score[i] = ez;                          // reuse buffer in place
  atomicAdd(&den[b * N_ + d], ez);
}

// ---------------- edge pass 3: out[dst] += (ez/denom[dst]) * xl[src] ----------------
__global__ void __launch_bounds__(256) edge_agg_k(const float* __restrict__ ez,
                                                  const float* __restrict__ den,
                                                  const float* __restrict__ XL,
                                                  const int* __restrict__ ei,
                                                  float* __restrict__ out) {
  int wave = threadIdx.x >> 5, lane = threadIdx.x & 31;
  int w = blockIdx.x * 8 + wave;
  if (w >= EDG_) return;
  int b = w / ET_, e = w % ET_;
  int s, d;
  if (e < E_) { s = ei[b * 2 * E_ + e]; d = ei[b * 2 * E_ + E_ + e]; }
  else        { s = d = e - E_; }
  float a = ez[w] / den[b * N_ + d];
  const float* pl = XL + (b * N_ + s) * H_;
  float* po = out + (b * N_ + d) * H_;
#pragma unroll
  for (int h = lane; h < H_; h += 32) atomicAdd(&po[h], a * pl[h]);
}

// ---------------- finalize: x = relu(agg + gat_bias [+ sk]) (in place) ----------------
__global__ void __launch_bounds__(256) fin_k(float* __restrict__ buf,
                                             const float* __restrict__ gb,
                                             const float* __restrict__ sk) {
  int i = blockIdx.x * 256 + threadIdx.x;
  if (i >= NTH_) return;
  int h = i & 63;
  float v = buf[i] + gb[h];
  if (sk) v += sk[i];
  buf[i] = fmaxf(v, 0.f);
}

// ---------------- heads ----------------
__global__ void __launch_bounds__(256) logits_k(const float* __restrict__ x3,
                                                const float* __restrict__ wlb,
                                                const float* __restrict__ blb,
                                                float* __restrict__ o2) {
  int i = blockIdx.x * 256 + threadIdx.x;
  if (i >= NN_ * 4) return;
  int j = i & 3, bn = i >> 2;
  const float* xp = x3 + bn * H_;
  float acc = blb[j];
  for (int k = 0; k < H_; ++k) acc += xp[k] * wlb[k * 4 + j];
  o2[i] = acc;
}
__global__ void __launch_bounds__(256) value_k(const float* __restrict__ x4,
                                               const float* __restrict__ wvl,
                                               const float* __restrict__ bvl,
                                               float* __restrict__ o3) {
  int i = blockIdx.x * 256 + threadIdx.x;
  if (i >= NN_) return;
  const float* xp = x4 + i * H_;
  float acc = bvl[0];
  for (int k = 0; k < H_; ++k) acc += xp[k] * wvl[k];
  o3[i] = acc;
}

// ---------------- host side ----------------
static void run_edge_phase(hipStream_t s, const int* ei, const float* att,
                           float* XL, float* XR, unsigned int* MB, float* DEN,
                           float* SCORE, float* OUTBUF, const float* gb, const float* SK) {
  const int EWB = EDG_ / 8;                       // wave-per-edge blocks (exact)
  const int ETB = (EDG_ + 255) / 256;             // thread-per-edge blocks
  initnd_k   <<<NN_ / 256, 256, 0, s>>>(MB, DEN);
  edge_score_k<<<EWB,       256, 0, s>>>(XL, XR, ei, att, SCORE, MB);
  edge_soft_k<<<ETB,        256, 0, s>>>(SCORE, MB, DEN, ei);
  zero_k     <<<NTH_ / 256, 256, 0, s>>>(OUTBUF, NTH_);
  edge_agg_k <<<EWB,        256, 0, s>>>(SCORE, DEN, XL, ei, OUTBUF);
  fin_k      <<<NTH_ / 256, 256, 0, s>>>(OUTBUF, gb, SK);
}

extern "C" void kernel_launch(void* const* d_in, const int* in_sizes, int n_in,
                              void* d_out, int out_size, void* d_ws, size_t ws_size,
                              hipStream_t stream) {
  const float* obs = (const float*)d_in[0];
  const int*   ei  = (const int*)  d_in[1];
  const float* w1  = (const float*)d_in[2];  const float* b1  = (const float*)d_in[3];
  const float* w2  = (const float*)d_in[4];  const float* b2  = (const float*)d_in[5];
  const float* w3  = (const float*)d_in[6];  const float* b3  = (const float*)d_in[7];
  // gcn params: Wl, bl, Wr, br, att, bias
  const float* g1[6]; const float* g2[6]; const float* g3[6]; const float* g4[6];
  for (int k = 0; k < 6; ++k) {
    g1[k] = (const float*)d_in[8  + k];
    g2[k] = (const float*)d_in[14 + k];
    g3[k] = (const float*)d_in[20 + k];
    g4[k] = (const float*)d_in[26 + k];
  }
  const float* wsk = (const float*)d_in[32]; const float* bsk = (const float*)d_in[33];
  const float* wlb = (const float*)d_in[34]; const float* blb = (const float*)d_in[35];
  const float* wvl = (const float*)d_in[36]; const float* bvl = (const float*)d_in[37];

  float* out = (float*)d_out;
  float* o0 = out;                       // obs[:,:,:4]   (B*N*4)
  float* o1 = out + 262144;              // obs[:,:,4:5]  (B*N)
  float* o2 = out + 327680;              // logits        (B*N*4)
  float* o3 = out + 589824;              // values        (B*N)
  float* o4 = out + 655360;              // latent        (B*N*3)

  // workspace layout (floats)
  const size_t F  = (size_t)NTH_;
  const size_t L3 = (size_t)NN_ * 3;
  float* ws   = (float*)d_ws;
  float* XL   = ws;
  float* XR   = ws + F;
  float* BA   = ws + 2 * F;              // h1 / x1 / x3
  float* BB   = ws + 3 * F;              // h2 / x2
  float* BC   = ws + 4 * F;              // x4
  float* SK   = ws + 5 * F;
  float* LAT  = ws + 6 * F;
  float* XG   = LAT + L3;
  unsigned int* MB = (unsigned int*)(XG + NN_);
  float* DEN  = ws + 6 * F + L3 + 2 * NN_;
  float* SCORE= ws + 6 * F + L3 + 3 * NN_;

  const int GB = TILES_ / 8;             // wmma gemm blocks (exact, full waves)

  // pass-through outputs
  copy_obs_k<<<NN_ / 256, 256, 0, stream>>>(obs, o0, o1);

  // encoder
  enc1_k  <<<NTH_ / 256, 256, 0, stream>>>(obs, w1, b1, BA);
  gemm64_k<<<GB,         256, 0, stream>>>(BA, w2, b2, BB, 1);
  enc3_k  <<<NN_ / 256,  256, 0, stream>>>(BB, w3, b3, LAT, o4, XG);

  // skip projection
  skip_k<<<NTH_ / 256, 256, 0, stream>>>(LAT, wsk, bsk, SK);

  // GAT layer 1 (fin = 1) -> x1 in BA
  lin1_k<<<NTH_ / 256, 256, 0, stream>>>(XG, g1[0], g1[1], g1[2], g1[3], XL, XR);
  run_edge_phase(stream, ei, g1[4], XL, XR, MB, DEN, SCORE, BA, g1[5], nullptr);

  // GAT layer 2 -> x2 in BB
  xlxr64_k<<<GB, 256, 0, stream>>>(BA, g2[0], g2[1], g2[2], g2[3], XL, XR);
  run_edge_phase(stream, ei, g2[4], XL, XR, MB, DEN, SCORE, BB, g2[5], nullptr);

  // GAT layer 3 -> x3 in BA ; logits head
  xlxr64_k<<<GB, 256, 0, stream>>>(BB, g3[0], g3[1], g3[2], g3[3], XL, XR);
  run_edge_phase(stream, ei, g3[4], XL, XR, MB, DEN, SCORE, BA, g3[5], SK);
  logits_k<<<(NN_ * 4) / 256, 256, 0, stream>>>(BA, wlb, blb, o2);

  // GAT layer 4 -> x4 in BC ; value head
  xlxr64_k<<<GB, 256, 0, stream>>>(BB, g4[0], g4[1], g4[2], g4[3], XL, XR);
  run_edge_phase(stream, ei, g4[4], XL, XR, MB, DEN, SCORE, BC, g4[5], SK);
  value_k<<<NN_ / 256, 256, 0, stream>>>(BC, wvl, bvl, o3);
}
